// MultilayerGRU_76029511074313
// MI455X (gfx1250) — matrix operand order repaired and checked by reading, not compiled, not run on hardware
//
#include <hip/hip_runtime.h>

typedef __attribute__((ext_vector_type(16))) __bf16 v16bf;
typedef __attribute__((ext_vector_type(8)))  __bf16 v8bf;
typedef __attribute__((ext_vector_type(8)))  float  v8f;

// global-address-space pointer types: keep weight streaming on global_load_b128
// (LOADcnt only) instead of flat_load_b128 (LOADcnt+DScnt, couples with LDS waits)
typedef const __attribute__((address_space(1))) unsigned short* gwptr;
typedef const __attribute__((address_space(1))) v8bf*           gv8p;

#define B_  32
#define S_  2048
#define I_  128
#define H_  512
#define O_  128
#define SA  520   // padded bf16 row stride for H=512 A-buffers (avoids LDS bank conflicts)
#define SAX 136   // padded bf16 row stride for I=128 A-buffer

// bf16 weight layout offsets inside d_ws (units: halves)
static constexpr size_t OFF_L0_WXZ = 0;
static constexpr size_t OFF_L0_WXR = OFF_L0_WXZ + 512 * 128;
static constexpr size_t OFF_L0_WXG = OFF_L0_WXR + 512 * 128;
static constexpr size_t OFF_L0_WHZ = OFF_L0_WXG + 512 * 128;
static constexpr size_t OFF_L0_WHR = OFF_L0_WHZ + 512 * 512;
static constexpr size_t OFF_L0_WHG = OFF_L0_WHR + 512 * 512;
static constexpr size_t OFF_L1_WXZ = OFF_L0_WHG + 512 * 512;
static constexpr size_t OFF_L1_WHZ = OFF_L1_WXZ + 512 * 512;
static constexpr size_t OFF_L1_WXR = OFF_L1_WHZ + 512 * 512;
static constexpr size_t OFF_L1_WHR = OFF_L1_WXR + 512 * 512;
static constexpr size_t OFF_L1_WXG = OFF_L1_WHR + 512 * 512;
static constexpr size_t OFF_L1_WHG = OFF_L1_WXG + 512 * 512;
static constexpr size_t OFF_WY     = OFF_L1_WHG + 512 * 512;

__device__ __forceinline__ unsigned short f2bf(float f) {
  union { float f; unsigned int u; } v; v.f = f;
  unsigned int u = v.u;
  unsigned int r = (u + 0x7FFFu + ((u >> 16) & 1u)) >> 16;  // round-to-nearest-even
  return (unsigned short)r;
}

__device__ __forceinline__ float sigmf(float x) {
  return 1.0f / (1.0f + __expf(-x));
}

// A fragment (16x32 bf16, ISA 16-bit A layout) from an LDS row-major buffer.
// lane<16: K pairs {0..7,16..23}; lane>=16: {8..15,24..31} -> two 16B chunks.
__device__ __forceinline__ v16bf load_afrag(const unsigned short* buf, int stride,
                                            int kt, int laneLo, int laneHi) {
  const v8bf* p = (const v8bf*)(buf + laneLo * stride + kt * 32 + laneHi * 8);
  v8bf lo = p[0];
  v8bf hi = p[2];   // +16 halves
  return __builtin_shufflevector(lo, hi, 0,1,2,3,4,5,6,7,8,9,10,11,12,13,14,15);
}

// B fragment (32x16 bf16): lane holds column N=lane%16 = row n of W[N][K];
// lanes 0-15 -> K {0..7,16..23}, lanes 16-31 -> K {8..15,24..31} (matches A).
__device__ __forceinline__ v16bf load_bfrag(gwptr W, int K, int n, int kt, int laneHi) {
  gv8p p = (gv8p)(W + (size_t)n * K + kt * 32 + laneHi * 8);
  return __builtin_shufflevector(p[0], p[2], 0,1,2,3,4,5,6,7,8,9,10,11,12,13,14,15);
}

__device__ __forceinline__ v8f wmma_bf16(v16bf a, v16bf b, v8f c) {
  return __builtin_amdgcn_wmma_f32_16x16x32_bf16(false, a, false, b, (short)0, c, false, false);
}

// acc[0..3] += A(16xK) x W^T tiles; wave owns N-tiles ntb..ntb+3.
// Software-pipelined: fragments for kt+1 are issued before the 4 WMMAs of kt,
// so global_load latency (L2-resident weights) overlaps with matrix work.
template <int K>
__device__ __forceinline__ void mm4(v8f* acc, const unsigned short* A, int strideA,
                                    gwptr W, int ntb, int laneLo, int laneHi) {
  constexpr int NKT = K / 32;
  const int n0 = (ntb + 0) * 16 + laneLo;
  const int n1 = (ntb + 1) * 16 + laneLo;
  const int n2 = (ntb + 2) * 16 + laneLo;
  const int n3 = (ntb + 3) * 16 + laneLo;

  v16bf a  = load_afrag(A, strideA, 0, laneLo, laneHi);
  v16bf b0 = load_bfrag(W, K, n0, 0, laneHi);
  v16bf b1 = load_bfrag(W, K, n1, 0, laneHi);
  v16bf b2 = load_bfrag(W, K, n2, 0, laneHi);
  v16bf b3 = load_bfrag(W, K, n3, 0, laneHi);

#pragma unroll
  for (int kt = 0; kt < NKT; ++kt) {
    v16bf an, c0, c1, c2, c3;
    if (kt + 1 < NKT) {  // issue next-slice loads before using current regs
      an = load_afrag(A, strideA, kt + 1, laneLo, laneHi);
      c0 = load_bfrag(W, K, n0, kt + 1, laneHi);
      c1 = load_bfrag(W, K, n1, kt + 1, laneHi);
      c2 = load_bfrag(W, K, n2, kt + 1, laneHi);
      c3 = load_bfrag(W, K, n3, kt + 1, laneHi);
    }
    acc[0] = wmma_bf16(a, b0, acc[0]);
    acc[1] = wmma_bf16(a, b1, acc[1]);
    acc[2] = wmma_bf16(a, b2, acc[2]);
    acc[3] = wmma_bf16(a, b3, acc[3]);
    if (kt + 1 < NKT) { a = an; b0 = c0; b1 = c1; b2 = c2; b3 = c3; }
  }
}

__global__ void cvt_bf16_kernel(const float* __restrict__ s, unsigned short* __restrict__ d, int n) {
  int i = blockIdx.x * blockDim.x + threadIdx.x;
  int stride = gridDim.x * blockDim.x;
  for (; i < n; i += stride) d[i] = f2bf(s[i]);
}

__global__ __launch_bounds__(256) void gru_persistent_kernel(
    const float* __restrict__ x, const float* __restrict__ h0i,
    const float* __restrict__ bxz0, const float* __restrict__ bxr0, const float* __restrict__ bxg0,
    const float* __restrict__ bxz1, const float* __restrict__ bxr1, const float* __restrict__ bxg1,
    const float* __restrict__ byv, const unsigned short* __restrict__ wbf,
    float* __restrict__ out) {
  __shared__ unsigned short A_x[16 * SAX];   // xt tile, bf16 (rows 2-15 zero)
  __shared__ unsigned short A_h0[16 * SA];   // layer0 hidden / layer1 input, bf16
  __shared__ unsigned short A_h1[16 * SA];   // layer1 hidden, bf16
  __shared__ unsigned short A_rh[16 * SA];   // r*h scratch (both layers), bf16
  __shared__ float h0f[2 * H_];              // fp32 state, 2 batch rows
  __shared__ float h1f[2 * H_];

  const int tid    = threadIdx.x;
  const int lane   = tid & 31;
  const int wave   = tid >> 5;        // 8 waves
  const int laneLo = lane & 15;
  const int laneHi = lane >> 4;       // 0|1
  const int bb     = blockIdx.x * 2;  // batch base (2 rows per WG)
  const int ntb    = wave * 4;        // this wave's N-tile base for H=512 gates

  // ---- init: zero A buffers (rows 2-15 must stay zero), load initial hidden ----
  for (int i = tid; i < 16 * SAX; i += 256) A_x[i] = 0;
  for (int i = tid; i < 16 * SA; i += 256) { A_h0[i] = 0; A_h1[i] = 0; A_rh[i] = 0; }
  for (int i = tid; i < 2 * H_; i += 256) {
    int b = i >> 9, c = i & (H_ - 1);
    float v0 = h0i[((size_t)(bb + b) * 2 + 0) * H_ + c];
    float v1 = h0i[((size_t)(bb + b) * 2 + 1) * H_ + c];
    h0f[i] = v0; h1f[i] = v1;
    A_h0[b * SA + c] = f2bf(v0);
    A_h1[b * SA + c] = f2bf(v1);
  }
  __syncthreads();

  const v8f vzero = {};
  float zk[4][2];

  for (int t = 0; t < S_; ++t) {
    // Launder the weight base pointer each iteration so LICM cannot hoist the
    // (loop-invariant) weight fragment loads out of the time loop (that caused
    // a 20KB/lane scratch spill storm). Cast to address_space(1) so the loads
    // lower to global_load_b128 (LOADcnt only), not flat_load (LOADcnt+DScnt).
    unsigned long long wq = (unsigned long long)wbf;
    asm volatile("" : "+s"(wq));
    gwptr wb = (gwptr)wq;
    gwptr Wl0xz = wb + OFF_L0_WXZ;
    gwptr Wl0xr = wb + OFF_L0_WXR;
    gwptr Wl0xg = wb + OFF_L0_WXG;
    gwptr Wl0hz = wb + OFF_L0_WHZ;
    gwptr Wl0hr = wb + OFF_L0_WHR;
    gwptr Wl0hg = wb + OFF_L0_WHG;
    gwptr Wl1xz = wb + OFF_L1_WXZ;
    gwptr Wl1hz = wb + OFF_L1_WHZ;
    gwptr Wl1xr = wb + OFF_L1_WXR;
    gwptr Wl1hr = wb + OFF_L1_WHR;
    gwptr Wl1xg = wb + OFF_L1_WXG;
    gwptr Wl1hg = wb + OFF_L1_WHG;
    gwptr Wy    = wb + OFF_WY;

    // ---- phase 1: stage xt (2 x 128 fp32 -> bf16 A tile) ----
    {
      int b = tid >> 7, i = tid & 127;
      A_x[b * SAX + i] = f2bf(x[((size_t)(bb + b) * S_ + t) * I_ + i]);
    }
    __syncthreads();

    // ---- phase 2: layer0 z, r preacts ----
    {
      v8f zc[4], rc[4];
#pragma unroll
      for (int j = 0; j < 4; ++j) { zc[j] = vzero; rc[j] = vzero; }
      mm4<128>(zc, A_x, SAX, Wl0xz, ntb, laneLo, laneHi);
      mm4<512>(zc, A_h0, SA, Wl0hz, ntb, laneLo, laneHi);
      mm4<128>(rc, A_x, SAX, Wl0xr, ntb, laneLo, laneHi);
      mm4<512>(rc, A_h0, SA, Wl0hr, ntb, laneLo, laneHi);
      if (laneHi == 0) {
#pragma unroll
        for (int j = 0; j < 4; ++j) {
          int col = wave * 64 + j * 16 + laneLo;
          float bz = bxz0[col], br = bxr0[col];
          float z0 = sigmf(zc[j][0] + bz), z1 = sigmf(zc[j][1] + bz);
          float r0 = sigmf(rc[j][0] + br), r1 = sigmf(rc[j][1] + br);
          zk[j][0] = z0; zk[j][1] = z1;
          A_rh[col]      = f2bf(r0 * h0f[col]);
          A_rh[SA + col] = f2bf(r1 * h0f[H_ + col]);
        }
      }
    }
    __syncthreads();

    // ---- phase 3: layer0 g, update h0 ----
    {
      v8f gc[4];
#pragma unroll
      for (int j = 0; j < 4; ++j) gc[j] = vzero;
      mm4<128>(gc, A_x, SAX, Wl0xg, ntb, laneLo, laneHi);
      mm4<512>(gc, A_rh, SA, Wl0hg, ntb, laneLo, laneHi);
      if (laneHi == 0) {
#pragma unroll
        for (int j = 0; j < 4; ++j) {
          int col = wave * 64 + j * 16 + laneLo;
          float bg = bxg0[col];
          float g0 = tanhf(gc[j][0] + bg), g1 = tanhf(gc[j][1] + bg);
          float n0 = zk[j][0] * h0f[col]      + (1.0f - zk[j][0]) * g0;
          float n1 = zk[j][1] * h0f[H_ + col] + (1.0f - zk[j][1]) * g1;
          h0f[col] = n0; h0f[H_ + col] = n1;
          A_h0[col] = f2bf(n0); A_h0[SA + col] = f2bf(n1);
        }
      }
    }
    __syncthreads();

    // ---- phase 4: layer1 z, r preacts (input = new h0) ----
    {
      v8f zc[4], rc[4];
#pragma unroll
      for (int j = 0; j < 4; ++j) { zc[j] = vzero; rc[j] = vzero; }
      mm4<512>(zc, A_h0, SA, Wl1xz, ntb, laneLo, laneHi);
      mm4<512>(zc, A_h1, SA, Wl1hz, ntb, laneLo, laneHi);
      mm4<512>(rc, A_h0, SA, Wl1xr, ntb, laneLo, laneHi);
      mm4<512>(rc, A_h1, SA, Wl1hr, ntb, laneLo, laneHi);
      if (laneHi == 0) {
#pragma unroll
        for (int j = 0; j < 4; ++j) {
          int col = wave * 64 + j * 16 + laneLo;
          float bz = bxz1[col], br = bxr1[col];
          float z0 = sigmf(zc[j][0] + bz), z1 = sigmf(zc[j][1] + bz);
          float r0 = sigmf(rc[j][0] + br), r1 = sigmf(rc[j][1] + br);
          zk[j][0] = z0; zk[j][1] = z1;
          A_rh[col]      = f2bf(r0 * h1f[col]);
          A_rh[SA + col] = f2bf(r1 * h1f[H_ + col]);
        }
      }
    }
    __syncthreads();

    // ---- phase 5: layer1 g, update h1 ----
    {
      v8f gc[4];
#pragma unroll
      for (int j = 0; j < 4; ++j) gc[j] = vzero;
      mm4<512>(gc, A_h0, SA, Wl1xg, ntb, laneLo, laneHi);
      mm4<512>(gc, A_rh, SA, Wl1hg, ntb, laneLo, laneHi);
      if (laneHi == 0) {
#pragma unroll
        for (int j = 0; j < 4; ++j) {
          int col = wave * 64 + j * 16 + laneLo;
          float bg = bxg1[col];
          float g0 = tanhf(gc[j][0] + bg), g1 = tanhf(gc[j][1] + bg);
          float n0 = zk[j][0] * h1f[col]      + (1.0f - zk[j][0]) * g0;
          float n1 = zk[j][1] * h1f[H_ + col] + (1.0f - zk[j][1]) * g1;
          h1f[col] = n0; h1f[H_ + col] = n1;
          A_h1[col] = f2bf(n0); A_h1[SA + col] = f2bf(n1);
        }
      }
    }
    __syncthreads();

    // ---- phase 6: y = h1 @ wy^T + by (N=128 -> 1 tile per wave) ----
    {
      const int ny = wave * 16 + laneLo;
      v8f yc = vzero;
      v16bf a = load_afrag(A_h1, SA, 0, laneLo, laneHi);
      v16bf b = load_bfrag(Wy, 512, ny, 0, laneHi);
#pragma unroll
      for (int kt = 0; kt < 16; ++kt) {
        v16bf an, bn;
        if (kt + 1 < 16) {
          an = load_afrag(A_h1, SA, kt + 1, laneLo, laneHi);
          bn = load_bfrag(Wy, 512, ny, kt + 1, laneHi);
        }
        yc = wmma_bf16(a, b, yc);
        if (kt + 1 < 16) { a = an; b = bn; }
      }
      if (laneHi == 0) {
        int col = wave * 16 + laneLo;
        float bo = byv[col];
        out[((size_t)bb * S_ + t) * O_ + col]       = yc[0] + bo;
        out[((size_t)(bb + 1) * S_ + t) * O_ + col] = yc[1] + bo;
      }
    }
    // no barrier needed: next phase 1 only writes A_x, barriered before phase 2
  }

  __syncthreads();
  // ---- final hidden state: out[B*S*O + ((b*2 + layer)*H + c)] ----
  const size_t HO = (size_t)B_ * S_ * O_;
  for (int i = tid; i < 2 * 2 * H_; i += 256) {
    int b = i >> 10, layer = (i >> 9) & 1, c = i & (H_ - 1);
    float v = layer ? h1f[b * H_ + c] : h0f[b * H_ + c];
    out[HO + ((size_t)(bb + b) * 2 + layer) * H_ + c] = v;
  }
}

extern "C" void kernel_launch(void* const* d_in, const int* in_sizes, int n_in,
                              void* d_out, int out_size, void* d_ws, size_t ws_size,
                              hipStream_t stream) {
  (void)in_sizes; (void)n_in; (void)out_size; (void)ws_size;
  unsigned short* wbf = (unsigned short*)d_ws;

  struct Cvt { int idx; size_t off; int n; };
  const Cvt cv[13] = {
      {2,  OFF_L0_WXZ, 512 * 128}, {5,  OFF_L0_WXR, 512 * 128}, {8,  OFF_L0_WXG, 512 * 128},
      {4,  OFF_L0_WHZ, 512 * 512}, {7,  OFF_L0_WHR, 512 * 512}, {10, OFF_L0_WHG, 512 * 512},
      {11, OFF_L1_WXZ, 512 * 512}, {13, OFF_L1_WHZ, 512 * 512}, {14, OFF_L1_WXR, 512 * 512},
      {16, OFF_L1_WHR, 512 * 512}, {17, OFF_L1_WXG, 512 * 512}, {19, OFF_L1_WHG, 512 * 512},
      {20, OFF_WY,     128 * 512}};
  for (int i = 0; i < 13; ++i) {
    int blocks = (cv[i].n + 255) / 256;
    if (blocks > 512) blocks = 512;
    cvt_bf16_kernel<<<blocks, 256, 0, stream>>>((const float*)d_in[cv[i].idx],
                                                wbf + cv[i].off, cv[i].n);
  }

  gru_persistent_kernel<<<16, 256, 0, stream>>>(
      (const float*)d_in[0],  // x
      (const float*)d_in[1],  // h0
      (const float*)d_in[3],  // l0_bxz
      (const float*)d_in[6],  // l0_bxr
      (const float*)d_in[9],  // l0_bxg
      (const float*)d_in[12], // l1_bxz
      (const float*)d_in[15], // l1_bxr
      (const float*)d_in[18], // l1_bxg
      (const float*)d_in[21], // by
      wbf, (float*)d_out);
}
